// BasicRNN_39676907888252
// MI455X (gfx1250) — matrix operand order repaired and checked by reading, not compiled, not run on hardware
//
#include <hip/hip_runtime.h>

typedef __attribute__((ext_vector_type(16))) _Float16 v16h;
typedef __attribute__((ext_vector_type(8)))  float    v8f;

#define T_STEPS 1024
#define BATCH   64
#define I_DIM   256
#define H_DIM   512
#define O_DIM   256

#define XW_BYTES ((size_t)T_STEPS * BATCH * H_DIM * 4)   // 128 MB f32
#define HS_BYTES ((size_t)T_STEPS * BATCH * H_DIM * 2)   //  64 MB f16

// Fast hyperbolic tangent: prefer the CDNA5 V_TANH_F32 transcendental if the
// toolchain exposes it; fall back to libm otherwise.
#if __has_builtin(__builtin_amdgcn_tanhf)
#define fast_tanh(x) __builtin_amdgcn_tanhf(x)
#elif __has_builtin(__builtin_amdgcn_tanh_f32)
#define fast_tanh(x) __builtin_amdgcn_tanh_f32(x)
#else
#define fast_tanh(x) tanhf(x)
#endif

// ---- WMMA helpers (CDNA5 16x16x32 f16 -> f32) --------------------------------

static __device__ __forceinline__ v8f wmma_f16(v16h a, v16h b, v8f c) {
  // 8 args: (neg_a, A, neg_b, B, c_mod, C, reuse_a, reuse_b)
  return __builtin_amdgcn_wmma_f32_16x16x32_f16(false, a, false, b, (short)0, c,
                                                false, false);
}

// A (16x32 f16), row-major f32 source: halves[i] = A[row][k0 + 16*hi + i]
static __device__ __forceinline__ v16h load_a_f32(const float* base, int row, int ld,
                                                  int k0, int hi) {
  const float* q0 = base + (size_t)row * ld + k0 + 16 * hi;
  const float* q1 = q0 + 8;
  v16h a;
#pragma unroll
  for (int i = 0; i < 8; ++i) { a[i] = (_Float16)q0[i]; a[i + 8] = (_Float16)q1[i]; }
  return a;
}

// A from f16 source (global hs or LDS h buffer)
static __device__ __forceinline__ v16h load_a_f16(const _Float16* base, int row, int ld,
                                                  int k0, int hi) {
  const _Float16* q0 = base + (size_t)row * ld + k0 + 16 * hi;
  const _Float16* q1 = q0 + 8;
  v16h a;
#pragma unroll
  for (int i = 0; i < 8; ++i) { a[i] = q0[i]; a[i + 8] = q1[i]; }
  return a;
}

// B (32x16 f16): halves[i] = W[n][k0 + 16*hi + i], W row-major over k -> one 32B load
static __device__ __forceinline__ v16h load_b_f16(const _Float16* w, int n, int ld,
                                                  int k0, int hi) {
  return *(const v16h*)(w + (size_t)n * ld + k0 + 16 * hi);
}

// ---- Phase 0: weights f32 -> f16 (same layout; k contiguous = B-operand ready) --

__global__ void k_convert(const float* __restrict__ wih, const float* __restrict__ whh,
                          const float* __restrict__ wfc, _Float16* __restrict__ wih16,
                          _Float16* __restrict__ whh16, _Float16* __restrict__ wfc16) {
  int idx = blockIdx.x * blockDim.x + threadIdx.x;
  if (idx < H_DIM * I_DIM) wih16[idx] = (_Float16)wih[idx];
  if (idx < H_DIM * H_DIM) whh16[idx] = (_Float16)whh[idx];
  if (idx < O_DIM * H_DIM) wfc16[idx] = (_Float16)wfc[idx];
}

// ---- Phase 1: xw[t,b,:] = x[b,t,:] @ W_ih^T + (b_ih + b_hh) ---------------------
// grid.x = (T*B)/16 = 4096 M-tiles; 256 threads = 8 waves, 4 N-tiles each (full H)

__global__ void __launch_bounds__(256) k_xproj(const float* __restrict__ x,
                                               const _Float16* __restrict__ wih16,
                                               const float* __restrict__ b_ih,
                                               const float* __restrict__ b_hh,
                                               float* __restrict__ xw) {
  const int lane = threadIdx.x & 31;
  const int wave = threadIdx.x >> 5;
  const int hi = lane >> 4;
  const int ln = lane & 15;
  const int m0 = blockIdx.x * 16;     // row in (T,B) flattening, B inner
  const int t  = m0 >> 6;
  const int b0 = m0 & 63;

  v8f c[4];
#pragma unroll
  for (int j = 0; j < 4; ++j) {
    const int n0 = wave * 64 + j * 16;
    const float bias = b_ih[n0 + ln] + b_hh[n0 + ln];
#pragma unroll
    for (int r = 0; r < 8; ++r) c[j][r] = bias;
  }

  const float* arow = x + ((size_t)b0 * T_STEPS + t) * I_DIM;  // + row*T*I below

  // software-pipelined B (and A) so loads overlap the WMMA chain
  v16h bcur[4];
#pragma unroll
  for (int j = 0; j < 4; ++j)
    bcur[j] = load_b_f16(wih16, wave * 64 + j * 16 + ln, I_DIM, 0, hi);
  v16h acur = load_a_f32(arow, ln, T_STEPS * I_DIM, 0, hi);

#pragma unroll
  for (int k0 = 0; k0 < I_DIM; k0 += 32) {
    v16h anxt = acur;
    v16h bnxt[4];
#pragma unroll
    for (int j = 0; j < 4; ++j) bnxt[j] = bcur[j];
    if (k0 + 32 < I_DIM) {
      anxt = load_a_f32(arow, ln, T_STEPS * I_DIM, k0 + 32, hi);
#pragma unroll
      for (int j = 0; j < 4; ++j)
        bnxt[j] = load_b_f16(wih16, wave * 64 + j * 16 + ln, I_DIM, k0 + 32, hi);
    }
#pragma unroll
    for (int j = 0; j < 4; ++j) c[j] = wmma_f16(acur, bcur[j], c[j]);
    acur = anxt;
#pragma unroll
    for (int j = 0; j < 4; ++j) bcur[j] = bnxt[j];
  }

#pragma unroll
  for (int j = 0; j < 4; ++j) {
    const int n0 = wave * 64 + j * 16;
#pragma unroll
    for (int r = 0; r < 8; ++r)
      xw[((size_t)m0 + r + 8 * hi) * H_DIM + n0 + ln] = c[j][r];
  }
}

// ---- Phase 2: serial scan. One WG (= one WGP) per 16-row batch slice. ----------
// 1024 threads = 32 waves, one 16-col N-tile each (covers H=512).
// CDNA5 LDS (320 KB/WGP) holds: h double-buffered f16 (32 KB) + the K<256 half of
// W_hh f16 (256 KB). The K>=256 half streams from L2 with 2-deep register
// pipelining issued before the LDS half so its latency hides under ds-fed WMMAs.

#define KL (H_DIM / 2)  // 256: K range resident in LDS

__global__ void __launch_bounds__(1024) k_rnn(const float* __restrict__ xw,
                                              const float* __restrict__ h0,
                                              const _Float16* __restrict__ whh16,
                                              _Float16* __restrict__ hs,
                                              float* __restrict__ hfinal) {
  __shared__ __align__(32) _Float16 hA[2][16 * H_DIM];     //  32 KB
  __shared__ __align__(32) _Float16 whhL[H_DIM * KL];      // 256 KB: W_hh[n][k], k<KL

  const int tid  = threadIdx.x;
  const int lane = tid & 31;
  const int wave = tid >> 5;
  const int hi = lane >> 4;
  const int ln = lane & 15;
  const int b0 = blockIdx.x * 16;
  const int n0 = wave * 16;

  // stage h0 (f32 -> f16)
  for (int idx = tid; idx < 16 * H_DIM; idx += 1024) {
    const int r = idx >> 9, k = idx & (H_DIM - 1);
    hA[0][idx] = (_Float16)h0[(size_t)(b0 + r) * H_DIM + k];
  }
  // stage W_hh lower-K half into LDS (16B chunks)
  for (int cidx = tid; cidx < H_DIM * (KL / 8); cidx += 1024) {
    const int n = cidx / (KL / 8), kc = cidx % (KL / 8);
    *(uint4*)&whhL[n * KL + kc * 8] = *(const uint4*)&whh16[(size_t)n * H_DIM + kc * 8];
  }
  __syncthreads();

  int p = 0;
  for (int t = 0; t < T_STEPS; ++t) {
    const float* xwt = xw + ((size_t)t * BATCH + b0) * H_DIM;
    v8f c;
#pragma unroll
    for (int r = 0; r < 8; ++r) c[r] = xwt[(size_t)(r + 8 * hi) * H_DIM + n0 + ln];

    // issue first two upper-half B loads now; latency hides under the LDS half
    v16h bg0 = load_b_f16(whh16, n0 + ln, H_DIM, KL, hi);
    v16h bg1 = load_b_f16(whh16, n0 + ln, H_DIM, KL + 32, hi);

    // K < 256: B resident in LDS
#pragma unroll
    for (int k0 = 0; k0 < KL; k0 += 32) {
      v16h a = load_a_f16(&hA[p][0], ln, H_DIM, k0, hi);
      v16h b = *(const v16h*)(&whhL[(n0 + ln) * KL + k0 + 16 * hi]);
      c = wmma_f16(a, b, c);
    }
    // K >= 256: B from L2, 2-deep pipelined
#pragma unroll
    for (int k0 = KL; k0 < H_DIM; k0 += 32) {
      v16h bnew = bg1;
      if (k0 + 64 < H_DIM) bg1 = load_b_f16(whh16, n0 + ln, H_DIM, k0 + 64, hi);
      v16h a = load_a_f16(&hA[p][0], ln, H_DIM, k0, hi);
      c = wmma_f16(a, bg0, c);
      bg0 = bnew;
    }

    _Float16* dst = &hA[p ^ 1][0];
#pragma unroll
    for (int r = 0; r < 8; ++r) {
      const float v = fast_tanh(c[r]);
      const int rr = r + 8 * hi;
      const _Float16 hv = (_Float16)v;
      dst[rr * H_DIM + n0 + ln] = hv;
      hs[((size_t)t * BATCH + b0 + rr) * H_DIM + n0 + ln] = hv;
      if (t == T_STEPS - 1)
        hfinal[(size_t)(b0 + rr) * H_DIM + n0 + ln] = v;
    }
    if (t + 1 < T_STEPS) {  // pull next step's xw slice toward the WGP
      __builtin_prefetch(xw + ((size_t)(t + 1) * BATCH + b0) * H_DIM + tid * 8, 0, 0);
    }
    __syncthreads();
    p ^= 1;
  }
}

// ---- Phase 3: out[b,t,:] = hs[t,b,:] @ W_fc^T + b_fc ---------------------------
// grid.x = 4096 M-tiles; 256 threads = 8 waves, 2 N-tiles each (full O=256)

__global__ void __launch_bounds__(256) k_outproj(const _Float16* __restrict__ hs,
                                                 const _Float16* __restrict__ wfc16,
                                                 const float* __restrict__ b_fc,
                                                 float* __restrict__ out) {
  const int lane = threadIdx.x & 31;
  const int wave = threadIdx.x >> 5;
  const int hi = lane >> 4;
  const int ln = lane & 15;
  const int m0 = blockIdx.x * 16;
  const int t  = m0 >> 6;
  const int b0 = m0 & 63;

  v8f c[2];
#pragma unroll
  for (int j = 0; j < 2; ++j) {
    const int n0 = wave * 32 + j * 16;
    const float bias = b_fc[n0 + ln];
#pragma unroll
    for (int r = 0; r < 8; ++r) c[j][r] = bias;
  }

  const _Float16* arow = hs + ((size_t)t * BATCH + b0) * H_DIM;

  v16h bcur[2];
#pragma unroll
  for (int j = 0; j < 2; ++j)
    bcur[j] = load_b_f16(wfc16, wave * 32 + j * 16 + ln, H_DIM, 0, hi);
  v16h acur = load_a_f16(arow, ln, H_DIM, 0, hi);

#pragma unroll
  for (int k0 = 0; k0 < H_DIM; k0 += 32) {
    v16h anxt = acur;
    v16h bnxt[2];
#pragma unroll
    for (int j = 0; j < 2; ++j) bnxt[j] = bcur[j];
    if (k0 + 32 < H_DIM) {
      anxt = load_a_f16(arow, ln, H_DIM, k0 + 32, hi);
#pragma unroll
      for (int j = 0; j < 2; ++j)
        bnxt[j] = load_b_f16(wfc16, wave * 32 + j * 16 + ln, H_DIM, k0 + 32, hi);
    }
#pragma unroll
    for (int j = 0; j < 2; ++j) c[j] = wmma_f16(acur, bcur[j], c[j]);
    acur = anxt;
#pragma unroll
    for (int j = 0; j < 2; ++j) bcur[j] = bnxt[j];
  }

#pragma unroll
  for (int j = 0; j < 2; ++j) {
    const int n0 = wave * 32 + j * 16;
#pragma unroll
    for (int r = 0; r < 8; ++r)
      out[(size_t)(b0 + r + 8 * hi) * T_STEPS * O_DIM + (size_t)t * O_DIM + n0 + ln] =
          c[j][r];
  }
}

// ---- Launch --------------------------------------------------------------------

extern "C" void kernel_launch(void* const* d_in, const int* in_sizes, int n_in,
                              void* d_out, int out_size, void* d_ws, size_t ws_size,
                              hipStream_t stream) {
  const float* x    = (const float*)d_in[0];
  const float* h0   = (const float*)d_in[1];
  const float* wih  = (const float*)d_in[2];
  const float* whh  = (const float*)d_in[3];
  const float* b_ih = (const float*)d_in[4];
  const float* b_hh = (const float*)d_in[5];
  const float* wfc  = (const float*)d_in[6];
  const float* b_fc = (const float*)d_in[7];

  char* ws = (char*)d_ws;
  float*    xw    = (float*)ws;                                   // (T,B,H) f32
  _Float16* hs    = (_Float16*)(ws + XW_BYTES);                   // (T,B,H) f16
  _Float16* wih16 = (_Float16*)(ws + XW_BYTES + HS_BYTES);        // (H,I)
  _Float16* whh16 = wih16 + (size_t)H_DIM * I_DIM;                // (H,H)
  _Float16* wfc16 = whh16 + (size_t)H_DIM * H_DIM;                // (O,H)

  float* out    = (float*)d_out;                                  // (B,T,O)
  float* hfinal = out + (size_t)BATCH * T_STEPS * O_DIM;          // (B,H)

  k_convert<<<(H_DIM * H_DIM + 255) / 256, 256, 0, stream>>>(wih, whh, wfc, wih16,
                                                             whh16, wfc16);
  k_xproj<<<(T_STEPS * BATCH) / 16, 256, 0, stream>>>(x, wih16, b_ih, b_hh, xw);
  k_rnn<<<BATCH / 16, 1024, 0, stream>>>(xw, h0, whh16, hs, hfinal);
  k_outproj<<<(T_STEPS * BATCH) / 16, 256, 0, stream>>>(hs, wfc16, b_fc, out);
}
